// ELMKT_72078141161592
// MI455X (gfx1250) — matrix operand order repaired and checked by reading, not compiled, not run on hardware
//
#include <hip/hip_runtime.h>
#include <hip/hip_bf16.h>

// ELMKT forward for MI455X (gfx1250, wave32, WMMA).
// Shapes: B=32, S=128, D=128, C=256, NQ=4000.
//   k_crall  : q-matrix gather -> cr_all [B*S, C] f32            (ws)
//   k_buildX : gather/concat fuse input -> X [B*S, 5D] f16       (ws)
//   k_fuse   : L_all = X @ W_fuse + b_fuse, WMMA f16->f32.
//              B staged in LDS fragment-ready; all 8 B frags of a k-step
//              preloaded before the 8-WMMA chain.
//   k_rec    : 1 block per batch element; h[C,D] f32 resident in LDS for
//              all 127 steps; per-step 256x128x128 GEMM with WMMA.
//              Wf1 pre-permuted once into fragment layout; A fragments
//              loaded as float4 vectors from LDS h and converted to f16.

typedef __attribute__((ext_vector_type(16))) _Float16 v16h;
typedef __attribute__((ext_vector_type(8)))  _Float16 v8h;
typedef __attribute__((ext_vector_type(8)))  float    v8f;

#define B_  32
#define S_  128
#define D_  128
#define C_  256

__device__ __forceinline__ float sigm(float x) { return 1.0f / (1.0f + __expf(-x)); }

union AFrag { v16h v; v8h h8[2]; };

// ---------------------------------------------------------------- prep kernels
__global__ void k_crall(const int* __restrict__ qseq, const int* __restrict__ Qtab,
                        const float* __restrict__ lQ, float* __restrict__ cr_all) {
    int idx = blockIdx.x * blockDim.x + threadIdx.x;        // B*S*C
    if (idx >= B_ * S_ * C_) return;
    int c = idx & (C_ - 1);
    int bs = idx >> 8;
    int q = qseq[bs];
    float Qf = (float)Qtab[q * C_ + c];
    float Q1 = fminf(Qf + 0.3f / 0.95f, 1.0f);
    float sg = sigm(lQ[q * C_ + c]);
    cr_all[idx] = (sg * 0.95f + Qf * 0.05f - (1.0f - Qf) * 0.05f) * Q1;
}

__global__ void k_buildX(const int* __restrict__ qseq, const int* __restrict__ cor,
                         const int* __restrict__ ut, const int* __restrict__ nh,
                         const int* __restrict__ na,
                         const float* __restrict__ eq, const float* __restrict__ eut,
                         _Float16* __restrict__ X) {
    int idx = blockIdx.x * blockDim.x + threadIdx.x;        // B*S*640
    if (idx >= B_ * S_ * 5 * D_) return;
    int k = idx % (5 * D_);
    int bs = idx / (5 * D_);
    int seg = k >> 7, d = k & (D_ - 1);
    float v;
    if (seg == 0)      v = eq[qseq[bs] * D_ + d];
    else if (seg == 1) v = (float)cor[bs];
    else if (seg == 2) v = eut[na[bs] * D_ + d];
    else if (seg == 3) v = eut[nh[bs] * D_ + d];
    else               v = eut[ut[bs] * D_ + d];
    X[idx] = (_Float16)v;
}

// --------------------------------------------------------- fuse GEMM (WMMA)
// L_all[4096,128] = X[4096,640] @ W_fuse[640,128] + b_fuse
__global__ void k_fuse(const _Float16* __restrict__ X, const float* __restrict__ Wfu,
                       const float* __restrict__ bfu, float* __restrict__ L_all) {
    extern __shared__ char smem[];
    _Float16* sbf = (_Float16*)smem;        // [20*8 frags][32 lanes][16 halves]
    int tid = threadIdx.x;
    // stage W_fuse into fragment-ready layout (one time, f32 -> f16)
    for (int idx = tid; idx < 20 * 8 * 32 * 16; idx += 256) {
        int hh = idx & 15;
        int l  = (idx >> 4) & 31;
        int f  = idx >> 9;                  // 0..159
        int ks = f >> 3, n = f & 7;
        int K   = ks * 32 + ((hh >> 3) << 4) + ((l >> 4) & 1) * 8 + (hh & 7);
        int col = n * 16 + (l & 15);
        sbf[idx] = (_Float16)Wfu[K * D_ + col];
    }
    __syncthreads();

    int wid  = tid >> 5;
    int lane = tid & 31;
    int nl   = lane & 15;
    int kb8  = (lane < 16) ? 0 : 8;

    int mt = blockIdx.x * 8 + wid;          // 0..255
    int rowbase = mt * 16;

    v8f acc[8];
    const v8f z8 = {0.f, 0.f, 0.f, 0.f, 0.f, 0.f, 0.f, 0.f};
#pragma unroll
    for (int n = 0; n < 8; ++n) acc[n] = z8;

    for (int ks = 0; ks < 20; ++ks) {
        AFrag a;
        const _Float16* xr = X + (rowbase + nl) * (5 * D_) + ks * 32;
        a.h8[0] = *(const v8h*)(xr + kb8);        // K = kb8 .. kb8+7
        a.h8[1] = *(const v8h*)(xr + 16 + kb8);   // K = 16+kb8 .. +7
        // preload all 8 B fragments, then issue the WMMA chain
        v16h bfr[8];
#pragma unroll
        for (int n = 0; n < 8; ++n)
            bfr[n] = *(const v16h*)(sbf + (((ks * 8 + n) * 32 + lane) << 4));
#pragma unroll
        for (int n = 0; n < 8; ++n)
            acc[n] = __builtin_amdgcn_wmma_f32_16x16x32_f16(
                false, a.v, false, bfr[n], (short)0, acc[n], false, false);
    }
    int rowadd = (lane < 16) ? 0 : 8;
#pragma unroll
    for (int n = 0; n < 8; ++n) {
        int col = n * 16 + nl;
        float bb = bfu[col];
#pragma unroll
        for (int r = 0; r < 8; ++r)
            L_all[(rowbase + r + rowadd) * D_ + col] = acc[n][r] + bb;
    }
}

// ------------------------------------------------------- recurrent kernel
__global__ void k_rec(const float* __restrict__ cr_all, const float* __restrict__ L_all,
                      const float* __restrict__ eq, const float* __restrict__ eut,
                      const float* __restrict__ Wg, const float* __restrict__ bg,
                      const float* __restrict__ Wa, const float* __restrict__ ba,
                      const float* __restrict__ Wf, const float* __restrict__ bfo,
                      const float* __restrict__ Wp, const float* __restrict__ bp,
                      const float* __restrict__ h0,
                      const int* __restrict__ qseq, const int* __restrict__ itseq,
                      float* __restrict__ out) {
    extern __shared__ char smem[];
    float*    sh_h  = (float*)smem;                          // [256*128] f32 state
    _Float16* sh_bf = (_Float16*)(smem + C_ * D_ * 4);       // [4*8][32][16] f16 Wf1 frags
    float*    sv    = (float*)(smem + C_ * D_ * 4 + D_ * D_ * 2);
    float* sh_cr    = sv;           // 256
    float* sh_crn   = sv + 256;     // 256
    float* sh_gin   = sv + 512;     // 512
    float* sh_itv   = sv + 1024;    // 128 each below
    float* sh_lc    = sv + 1152;
    float* sh_lpre  = sv + 1280;
    float* sh_qe    = sv + 1408;
    float* sh_abs   = sv + 1536;
    float* sh_ft    = sv + 1664;
    float* sh_ht    = sv + 1792;
    float* sh_htacc = sv + 1920;
    float* sh_gain  = sv + 2048;
    float* sh_pred  = sv + 2176;

    const int tid  = threadIdx.x;
    const int wid  = tid >> 5;
    const int lane = tid & 31;
    const int nl   = lane & 15;
    const int kb8  = (lane < 16) ? 0 : 8;
    const int rowadd = (lane < 16) ? 0 : 8;
    const int b = blockIdx.x;

    // init: h = h0 (shared across batch); Wf1 permuted into fragment layout
    for (int i = tid; i < C_ * D_; i += 256) sh_h[i] = h0[i];
    for (int idx = tid; idx < 4 * 8 * 32 * 16; idx += 256) {
        int hh = idx & 15;
        int l  = (idx >> 4) & 31;
        int f  = idx >> 9;                  // 0..31
        int ks = f >> 3, n = f & 7;
        int K   = ks * 32 + ((hh >> 3) << 4) + ((l >> 4) & 1) * 8 + (hh & 7);
        int col = n * 16 + (l & 15);
        sh_bf[idx] = (_Float16)Wf[K * D_ + col];
    }
    if (tid < C_) sh_cr[tid] = cr_all[(b * S_ + 0) * C_ + tid];
    __syncthreads();
    if (tid < D_) {
        float s = 0.f;
        for (int c = 0; c < C_; ++c) s += sh_cr[c] * sh_h[c * D_ + tid];
        sh_ht[tid]   = s;
        sh_lpre[tid] = 0.f;
    }
    if (tid == 0) out[b * S_] = 0.f;
    __syncthreads();

    for (int t = 0; t < S_ - 1; ++t) {
        // ---- per-step gathers
        sh_cr[tid]  = cr_all[(b * S_ + t) * C_ + tid];
        sh_crn[tid] = cr_all[(b * S_ + t + 1) * C_ + tid];
        if (tid < D_) {
            sh_itv[tid]   = eut[itseq[b * S_ + t] * D_ + tid];
            sh_lc[tid]    = L_all[(b * S_ + t) * D_ + tid];
            sh_qe[tid]    = eq[qseq[b * S_ + t + 1] * D_ + tid];
            sh_htacc[tid] = 0.f;
        }
        __syncthreads();
        // g_in = [l_pre, it, l, h_tilde]
        sh_gin[tid]       = (tid < D_) ? sh_lpre[tid] : sh_itv[tid - D_];
        sh_gin[256 + tid] = (tid < D_) ? sh_lc[tid]   : sh_ht[tid - D_];
        __syncthreads();
        // ---- gain / absorb (small vec-mat, VALU)
        if (tid < D_) {
            float s = bg[tid];
            for (int k = 0; k < 4 * D_; ++k) s += sh_gin[k] * Wg[k * D_ + tid];
            sh_gain[tid] = tanhf(s);
        } else {
            int d = tid - D_;
            float s = ba[d];
            for (int k = 0; k < 4 * D_; ++k) s += sh_gin[k] * Wa[k * D_ + d];
            sh_abs[d] = sigm(s);
        }
        __syncthreads();
        if (tid < D_) sh_abs[tid] *= (sh_gain[tid] + 1.0f) * 0.5f;
        __syncthreads();
        // ---- forget-gate broadcast term: absorb@Wf2 + it@Wf3 + b_forget
        if (tid < D_) {
            float s = bfo[tid];
            for (int j = 0; j < D_; ++j) s += sh_abs[j] * Wf[(D_ + j) * D_ + tid];
            for (int j = 0; j < D_; ++j) s += sh_itv[j] * Wf[(2 * D_ + j) * D_ + tid];
            sh_ft[tid] = s;
        }
        __syncthreads();

        // ---- WMMA phase: gammaLogit = h @ Wf1; h = cr*absorb + sigm(.)*h
        // wave w owns M-tiles {w, w+8}; rows disjoint across waves.
        for (int mi = 0; mi < 2; ++mi) {
            int mt = wid + mi * 8;
            int rowbase = mt * 16;
            v8f acc[8];
            const v8f z8 = {0.f, 0.f, 0.f, 0.f, 0.f, 0.f, 0.f, 0.f};
#pragma unroll
            for (int n = 0; n < 8; ++n) acc[n] = z8;
#pragma unroll
            for (int ks = 0; ks < 4; ++ks) {
                // A fragment: two contiguous 8-float runs, vector loads + cvt
                const float* hr = sh_h + (rowbase + nl) * D_ + ks * 32;
                float4 r0 = *(const float4*)(hr + kb8);
                float4 r1 = *(const float4*)(hr + kb8 + 4);
                float4 r2 = *(const float4*)(hr + 16 + kb8);
                float4 r3 = *(const float4*)(hr + 16 + kb8 + 4);
                v16h a;
                a[0]  = (_Float16)r0.x; a[1]  = (_Float16)r0.y;
                a[2]  = (_Float16)r0.z; a[3]  = (_Float16)r0.w;
                a[4]  = (_Float16)r1.x; a[5]  = (_Float16)r1.y;
                a[6]  = (_Float16)r1.z; a[7]  = (_Float16)r1.w;
                a[8]  = (_Float16)r2.x; a[9]  = (_Float16)r2.y;
                a[10] = (_Float16)r2.z; a[11] = (_Float16)r2.w;
                a[12] = (_Float16)r3.x; a[13] = (_Float16)r3.y;
                a[14] = (_Float16)r3.z; a[15] = (_Float16)r3.w;
                // preload all 8 B fragments, then issue the WMMA chain
                v16h bfr[8];
#pragma unroll
                for (int n = 0; n < 8; ++n)
                    bfr[n] = *(const v16h*)(sh_bf + (((ks * 8 + n) * 32 + lane) << 4));
#pragma unroll
                for (int n = 0; n < 8; ++n)
                    acc[n] = __builtin_amdgcn_wmma_f32_16x16x32_f16(
                        false, a, false, bfr[n], (short)0, acc[n], false, false);
            }
#pragma unroll
            for (int n = 0; n < 8; ++n) {
                int col = n * 16 + nl;
                float ft = sh_ft[col];
                float ab = sh_abs[col];
                float part = 0.f;
#pragma unroll
                for (int r = 0; r < 8; ++r) {
                    int row = rowbase + r + rowadd;
                    float gamma = sigm(acc[n][r] + ft);
                    float hn = sh_cr[row] * ab + gamma * sh_h[row * D_ + col];
                    sh_h[row * D_ + col] = hn;
                    part += sh_crn[row] * hn;
                }
                atomicAdd(&sh_htacc[col], part);   // ds_add_f32
            }
        }
        __syncthreads();

        // ---- prediction head
        if (tid < D_) {
            float s = bp[tid];
            for (int j = 0; j < D_; ++j) s += sh_qe[j] * Wp[j * D_ + tid];
            for (int j = 0; j < D_; ++j) s += sh_htacc[j] * Wp[(D_ + j) * D_ + tid];
            sh_pred[tid] = sigm(s);
        }
        __syncthreads();
        if (tid == 0) {
            float s = 0.f;
            for (int j = 0; j < D_; ++j) s += sh_pred[j];
            out[b * S_ + t + 1] = s / (float)D_;
        }
        // carry: l_pre = l, h_tilde = new h_tilde
        if (tid < D_) {
            sh_lpre[tid] = sh_lc[tid];
            sh_ht[tid]   = sh_htacc[tid];
        }
        __syncthreads();
    }
}

// ---------------------------------------------------------------- launcher
extern "C" void kernel_launch(void* const* d_in, const int* in_sizes, int n_in,
                              void* d_out, int out_size, void* d_ws, size_t ws_size,
                              hipStream_t stream) {
    const int*   qseq = (const int*)d_in[0];
    const int*   cor  = (const int*)d_in[1];
    const int*   ut   = (const int*)d_in[2];
    const int*   itq  = (const int*)d_in[3];
    const int*   nh   = (const int*)d_in[4];
    const int*   na   = (const int*)d_in[5];
    const int*   Qtab = (const int*)d_in[6];
    const float* lQ   = (const float*)d_in[7];
    const float* eq   = (const float*)d_in[8];
    const float* eut  = (const float*)d_in[9];
    const float* Wfu  = (const float*)d_in[10];
    const float* bfu  = (const float*)d_in[11];
    const float* Wg   = (const float*)d_in[12];
    const float* bg   = (const float*)d_in[13];
    const float* Wa   = (const float*)d_in[14];
    const float* ba   = (const float*)d_in[15];
    const float* Wf   = (const float*)d_in[16];
    const float* bfo  = (const float*)d_in[17];
    const float* Wp   = (const float*)d_in[18];
    const float* bp   = (const float*)d_in[19];
    const float* h0   = (const float*)d_in[20];
    float* out = (float*)d_out;

    // workspace layout (needs ~11.3 MB)
    char* ws = (char*)d_ws;
    float*    cr_all = (float*)ws;                            //  4 MB
    float*    L_all  = (float*)(ws + 4194304);                //  2 MB
    _Float16* X      = (_Float16*)(ws + 6291456);             //  5 MB

    k_crall <<<dim3((B_ * S_ * C_ + 255) / 256), dim3(256), 0, stream>>>(qseq, Qtab, lQ, cr_all);
    k_buildX<<<dim3((B_ * S_ * 5 * D_ + 255) / 256), dim3(256), 0, stream>>>(
        qseq, cor, ut, nh, na, eq, eut, X);
    k_fuse  <<<dim3(32), dim3(256), 20 * 8 * 32 * 16 * sizeof(_Float16), stream>>>(
        X, Wfu, bfu, L_all);

    size_t rec_lds = (size_t)C_ * D_ * 4 + (size_t)D_ * D_ * 2 + 2304 * 4; // 173 KB
    k_rec   <<<dim3(B_), dim3(256), rec_lds, stream>>>(
        cr_all, L_all, eq, eut, Wg, bg, Wa, ba, Wf, bfo, Wp, bp, h0, qseq, itq, out);
}
